// EncoderBlock_39170101739810
// MI455X (gfx1250) — matrix operand order repaired and checked by reading, not compiled, not run on hardware
//
#include <hip/hip_runtime.h>
#include <stdint.h>
#include <stddef.h>

// ---------------- problem constants ----------------
#define DIM_   2048
#define NH_    16
#define NKV_   4
#define DH_    128
#define FFN_   8192
#define B_     2
#define L_     2048
#define NTOK_  (B_ * L_)          // 4096

typedef __bf16 bf16_t;
typedef __attribute__((ext_vector_type(16))) __bf16 v16bf;
typedef __attribute__((ext_vector_type(8)))  __bf16 v8bf;
typedef __attribute__((ext_vector_type(8)))  float  v8f;

#if defined(__HIP_DEVICE_COMPILE__) && __has_builtin(__builtin_amdgcn_tensor_load_to_lds)
#define USE_TDM 1
#else
#define USE_TDM 0
#endif

static __device__ inline v16bf mk16(v8bf lo, v8bf hi) {
  v16bf r;
#pragma unroll
  for (int i = 0; i < 8; ++i) { r[i] = lo[i]; r[i + 8] = hi[i]; }
  return r;
}

static __device__ inline v8f wmma_bf16(v16bf a, v16bf b, v8f c) {
  // D = A(16x32) * B(32x16) + C, f32 accumulate
  return __builtin_amdgcn_wmma_f32_16x16x32_bf16(false, a, false, b, (short)0, c,
                                                 false, false);
}

#if USE_TDM
typedef __attribute__((ext_vector_type(4))) unsigned int v4u;
typedef __attribute__((ext_vector_type(8))) int v8i_t;
typedef __attribute__((ext_vector_type(4))) int v4i_t;

// DMA a (rows x 32) bf16 tile (row stride = k_stride elems) into LDS.
// TDM padding: 16 DWORDs (one 64B row) then +4 DWORDs -> LDS row stride 40 bf16,
// matching the bank-conflict-free fragment layout used by the compute waves.
static __device__ inline void tdm_load_tile32(uint32_t lds_off, const void* gaddr,
                                              int rows, int k_stride) {
  const uint64_t ga = (uint64_t)(uintptr_t)gaddr;
  v4u g0;
  g0[0] = 1u;                                                 // count=1 (valid)
  g0[1] = lds_off;                                            // LDS byte address
  g0[2] = (uint32_t)ga;                                       // global_addr[31:0]
  g0[3] = (uint32_t)((ga >> 32) & 0x01FFFFFFu) | (2u << 30);  // addr[56:32]|type=2
  const uint32_t dim0 = 0x40000000u, dim1 = 0x40000000u;      // huge: no OOB zeroing
  const uint64_t st = (uint64_t)(uint32_t)k_stride;           // dim0 stride (elems)
  v8i_t g1;
  g1[0] = (int)(0x00010000u       // data_size = 1 -> 2 bytes
                | (1u << 20)      // pad_enable
                | (3u << 22)      // pad_interval: 16 DWORDs (64B = one tile row)
                | (3u << 25));    // pad_amount: 4 DWORDs (8 bf16)
  g1[1] = (int)((dim0 & 0xFFFFu) << 16);
  g1[2] = (int)((dim0 >> 16) | ((dim1 & 0xFFFFu) << 16));
  g1[3] = (int)((dim1 >> 16) | (32u << 16));                  // tile_dim0 = 32
  g1[4] = (int)((uint32_t)rows & 0xFFFFu);                    // tile_dim1 = rows
  g1[5] = (int)(uint32_t)st;
  g1[6] = (int)(uint32_t)(st >> 32);
  g1[7] = 0;
  v4i_t z4 = {};
#if __clang_major__ >= 23
  v8i_t z8 = {};
  __builtin_amdgcn_tensor_load_to_lds(g0, g1, z4, z4, z8, 0);
#else
  __builtin_amdgcn_tensor_load_to_lds(g0, g1, z4, z4, 0);
#endif
}
#endif  // USE_TDM

// ---------------- elementwise helpers ----------------
__global__ __launch_bounds__(256)
void rmsnorm_kernel(const float* __restrict__ x, const float* __restrict__ w,
                    bf16_t* __restrict__ out, int dim) {
  __shared__ float red[8];
  const int row = blockIdx.x;
  const float* xr = x + (size_t)row * dim;
  float s = 0.f;
  for (int i = threadIdx.x; i < dim; i += 256) { float v = xr[i]; s += v * v; }
#pragma unroll
  for (int o = 16; o > 0; o >>= 1) s += __shfl_xor(s, o, 32);
  if ((threadIdx.x & 31) == 0) red[threadIdx.x >> 5] = s;
  __syncthreads();
  if (threadIdx.x == 0) {
    float t = 0.f;
#pragma unroll
    for (int i = 0; i < 8; ++i) t += red[i];
    red[0] = t;
  }
  __syncthreads();
  const float inv = rsqrtf(red[0] / (float)dim + 1e-6f);
  bf16_t* orow = out + (size_t)row * dim;
  for (int i = threadIdx.x; i < dim; i += 256)
    orow[i] = (bf16_t)(xr[i] * inv * w[i]);
}

// Transpose-cast: w[K][N] f32  ->  wt[N][K] bf16  (weights become B^T, K-contig)
__global__ __launch_bounds__(256)
void transpose_cvt_kernel(const float* __restrict__ w, bf16_t* __restrict__ wt,
                          int K, int N) {
  __shared__ float t[32][33];
  const int n0 = blockIdx.x * 32;
  const int k0 = blockIdx.y * 32;
  const int c = threadIdx.x & 31;
  const int r0 = threadIdx.x >> 5;  // 0..7
#pragma unroll
  for (int i = 0; i < 4; ++i) {
    const int r = r0 + i * 8;
    t[r][c] = w[(size_t)(k0 + r) * N + n0 + c];
  }
  __syncthreads();
#pragma unroll
  for (int i = 0; i < 4; ++i) {
    const int r = r0 + i * 8;
    wt[(size_t)(n0 + r) * K + k0 + c] = (bf16_t)t[c][r];
  }
}

// RoPE in-place on bf16 tensor laid out [ntok][nheads*128]
__global__ __launch_bounds__(256)
void rope_kernel(bf16_t* __restrict__ t, int nheads, int ntok, int seqlen) {
  size_t idx = (size_t)blockIdx.x * blockDim.x + threadIdx.x;
  const size_t total = (size_t)ntok * nheads * 64;
  if (idx >= total) return;
  const int i = (int)(idx % 64);
  const int h = (int)((idx / 64) % nheads);
  const size_t tok = idx / ((size_t)64 * nheads);
  const int pos = (int)(tok % seqlen);
  const float inv_freq = __expf(-((float)(2 * i) / 128.f) * 9.210340372f);
  const float ang = (float)pos * inv_freq;
  float sn, cs;
  __sincosf(ang, &sn, &cs);
  bf16_t* p = t + tok * (size_t)(nheads * 128) + (size_t)h * 128;
  const float x1 = (float)p[i];
  const float x2 = (float)p[i + 64];
  p[i]      = (bf16_t)(x1 * cs - x2 * sn);
  p[i + 64] = (bf16_t)(x1 * sn + x2 * cs);
}

// v: [b][l][kvh*128+d]  ->  vt: [b][kvh][d][L]
__global__ __launch_bounds__(256)
void vtrans_kernel(const bf16_t* __restrict__ v, bf16_t* __restrict__ vt,
                   int ntok, int seqlen) {
  size_t idx = (size_t)blockIdx.x * blockDim.x + threadIdx.x;
  const size_t total = (size_t)ntok * NKV_ * DH_;
  if (idx >= total) return;
  const int d = (int)(idx % DH_);
  const int kvh = (int)((idx / DH_) % NKV_);
  const size_t tok = idx / (DH_ * NKV_);
  const size_t b = tok / seqlen, l = tok % seqlen;
  vt[(((b * NKV_ + kvh) * DH_) + d) * (size_t)seqlen + l] =
      v[tok * (size_t)(NKV_ * DH_) + (size_t)kvh * DH_ + d];
}

// ---------------- tiled bf16 WMMA GEMM (TDM double-buffered) ----------------
// C(MxN) = A(MxK, row-major) * B, with B given transposed: BT(NxK, row-major).
// EPI 0: out bf16 = acc
// EPI 1: out f32  = aux_f32[idx] + acc                (fused residual)
// EPI 2: out bf16 = silu(aux_bf16[idx]) * acc         (fused SwiGLU)
template <int EPI>
__global__ __launch_bounds__(256)
void gemm_bf16_kernel(const bf16_t* __restrict__ A, const bf16_t* __restrict__ BT,
                      const void* __restrict__ aux, void* __restrict__ out,
                      int M, int N, int K) {
  constexpr int BM = 128, BN = 256, BK = 32, LD = 40;
  __shared__ bf16_t As[2][BM * LD];  // [m][k], stride-40 padded
  __shared__ bf16_t Bs[2][BN * LD];  // [n][k], stride-40 padded

  const int tid = threadIdx.x;
  const int lane = tid & 31;
  const int wave = tid >> 5;       // 0..7
  const int wm = wave >> 2;        // 0..1 -> 64 rows
  const int wn = wave & 3;         // 0..3 -> 64 cols
  const int l16 = lane & 15;
  const int lh = lane >> 4;
  const int m0 = blockIdx.y * BM;
  const int n0 = blockIdx.x * BN;
  const int NT = K / BK;

  v8f acc[4][4];
#pragma unroll
  for (int mt = 0; mt < 4; ++mt)
#pragma unroll
    for (int nt = 0; nt < 4; ++nt) acc[mt][nt] = v8f{};

  const bf16_t* aTile = A + (size_t)m0 * K;
  const bf16_t* bTile = BT + (size_t)n0 * K;

#if USE_TDM
  const uint32_t ldsA[2] = {(uint32_t)(uintptr_t)&As[0][0],
                            (uint32_t)(uintptr_t)&As[1][0]};
  const uint32_t ldsB[2] = {(uint32_t)(uintptr_t)&Bs[0][0],
                            (uint32_t)(uintptr_t)&Bs[1][0]};
  if (tid < 32) {  // wave 0 drives the DMA pipeline
    tdm_load_tile32(ldsA[0], aTile, BM, K);
    tdm_load_tile32(ldsB[0], bTile, BN, K);
  }
#endif

  for (int it = 0; it < NT; ++it) {
    const int buf = it & 1;
#if USE_TDM
    __syncthreads();  // everyone done reading buf^1 from iteration it-1
    if (tid < 32) {
      if (it + 1 < NT) {
        tdm_load_tile32(ldsA[buf ^ 1], aTile + (size_t)(it + 1) * BK, BM, K);
        tdm_load_tile32(ldsB[buf ^ 1], bTile + (size_t)(it + 1) * BK, BN, K);
        __builtin_amdgcn_s_wait_tensorcnt(2);  // current pair done, next in flight
      } else {
        __builtin_amdgcn_s_wait_tensorcnt(0);
      }
    }
    __syncthreads();  // current tile visible to all waves
#else
    __syncthreads();
    {  // stage A tile: thread -> row tid>>1, 16 elems
      const int r = tid >> 1;
      const int c = (tid & 1) * 16;
      const bf16_t* src = aTile + (size_t)r * K + it * BK + c;
      *(v8bf*)&As[buf][r * LD + c]     = *(const v8bf*)src;
      *(v8bf*)&As[buf][r * LD + c + 8] = *(const v8bf*)(src + 8);
    }
    {  // stage B tile: thread -> row tid, 32 elems (BT is K-contiguous)
      const bf16_t* src = bTile + (size_t)tid * K + it * BK;
#pragma unroll
      for (int j = 0; j < 4; ++j)
        *(v8bf*)&Bs[buf][tid * LD + j * 8] = *(const v8bf*)(src + j * 8);
    }
    __syncthreads();
#endif

    v16bf af[4];
#pragma unroll
    for (int mt = 0; mt < 4; ++mt) {
      const bf16_t* p = &As[buf][(wm * 64 + mt * 16 + l16) * LD + lh * 8];
      af[mt] = mk16(*(const v8bf*)p, *(const v8bf*)(p + 16));
    }
    v16bf bfr[4];
#pragma unroll
    for (int nt = 0; nt < 4; ++nt) {
      const bf16_t* p = &Bs[buf][(wn * 64 + nt * 16 + l16) * LD + lh * 8];
      bfr[nt] = mk16(*(const v8bf*)p, *(const v8bf*)(p + 16));
    }
#pragma unroll
    for (int mt = 0; mt < 4; ++mt)
#pragma unroll
      for (int nt = 0; nt < 4; ++nt)
        acc[mt][nt] = wmma_bf16(af[mt], bfr[nt], acc[mt][nt]);
  }

  // epilogue; C layout: VGPR i -> row i + 8*lh, col = l16 (per 16x16 tile)
#pragma unroll
  for (int mt = 0; mt < 4; ++mt)
#pragma unroll
    for (int nt = 0; nt < 4; ++nt) {
      const int col = n0 + wn * 64 + nt * 16 + l16;
#pragma unroll
      for (int i = 0; i < 8; ++i) {
        const int row = m0 + wm * 64 + mt * 16 + i + 8 * lh;
        const size_t idx = (size_t)row * N + col;
        const float a = acc[mt][nt][i];
        if constexpr (EPI == 0) {
          ((bf16_t*)out)[idx] = (bf16_t)a;
        } else if constexpr (EPI == 1) {
          ((float*)out)[idx] = ((const float*)aux)[idx] + a;
        } else {
          const float g = (float)((const bf16_t*)aux)[idx];
          const float s = g / (1.f + __expf(-g));
          ((bf16_t*)out)[idx] = (bf16_t)(s * a);
        }
      }
    }
}

// ---------------- flash attention (one wave per 16-query tile) ----------------
__global__ __launch_bounds__(32)
void attn_kernel(const bf16_t* __restrict__ q, const bf16_t* __restrict__ k,
                 const bf16_t* __restrict__ vt, bf16_t* __restrict__ out,
                 int seqlen) {
  constexpr int PLD = 40;
  __shared__ bf16_t P[16 * PLD];

  const int nqt = seqlen / 16;
  const int qt = blockIdx.x % nqt;
  const int h = (blockIdx.x / nqt) % NH_;
  const int b = blockIdx.x / (nqt * NH_);
  const int kvh = h >> 2;

  const int lane = threadIdx.x;
  const int l16 = lane & 15;
  const int lh = lane >> 4;
  const size_t qrow0 = (size_t)b * seqlen + (size_t)qt * 16;

  v16bf qf[4];
  {
    const bf16_t* qp = q + (qrow0 + l16) * (size_t)(NH_ * DH_) + (size_t)h * DH_;
#pragma unroll
    for (int ch = 0; ch < 4; ++ch) {
      const bf16_t* p = qp + ch * 32 + lh * 8;
      qf[ch] = mk16(*(const v8bf*)p, *(const v8bf*)(p + 16));
    }
  }

  v8f o[8];
#pragma unroll
  for (int t = 0; t < 8; ++t) o[t] = v8f{};
  float m[8], lsum[8];
#pragma unroll
  for (int i = 0; i < 8; ++i) { m[i] = -1e30f; lsum[i] = 0.f; }

  const float scale = 0.08838834764831845f;  // 1/sqrt(128)
  const int kend = (qt + 1) * 16;

  for (int kb = 0; kb < kend; kb += 32) {
    v8f s[2] = {v8f{}, v8f{}};
#pragma unroll
    for (int g = 0; g < 2; ++g) {
#pragma unroll
      for (int ch = 0; ch < 4; ++ch) {
        const bf16_t* p = k + ((size_t)b * seqlen + kb + g * 16 + l16) *
                                  (size_t)(NKV_ * DH_) +
                          (size_t)kvh * DH_ + ch * 32 + lh * 8;
        s[g] = wmma_bf16(qf[ch], mk16(*(const v8bf*)p, *(const v8bf*)(p + 16)),
                         s[g]);
      }
    }
#pragma unroll
    for (int g = 0; g < 2; ++g) {
      const int key = kb + g * 16 + l16;
#pragma unroll
      for (int i = 0; i < 8; ++i) {
        const int qrow = qt * 16 + i + 8 * lh;
        const float v = s[g][i] * scale;
        s[g][i] = (key > qrow) ? -1e30f : v;
      }
    }
#pragma unroll
    for (int i = 0; i < 8; ++i) {
      float mx = fmaxf(s[0][i], s[1][i]);
#pragma unroll
      for (int off = 8; off > 0; off >>= 1) mx = fmaxf(mx, __shfl_xor(mx, off, 32));
      const float mnew = fmaxf(m[i], mx);
      const float alpha = __expf(m[i] - mnew);
      m[i] = mnew;
      const float p0 = __expf(s[0][i] - mnew);
      const float p1 = __expf(s[1][i] - mnew);
      s[0][i] = p0; s[1][i] = p1;
      float rs = p0 + p1;
#pragma unroll
      for (int off = 8; off > 0; off >>= 1) rs += __shfl_xor(rs, off, 32);
      lsum[i] = lsum[i] * alpha + rs;
#pragma unroll
      for (int t = 0; t < 8; ++t) o[t][i] *= alpha;
    }
#pragma unroll
    for (int i = 0; i < 8; ++i) {
      const int r = i + 8 * lh;
      P[r * PLD + l16]      = (bf16_t)s[0][i];
      P[r * PLD + 16 + l16] = (bf16_t)s[1][i];
    }
    __syncthreads();
    v16bf pf;
    {
      const bf16_t* p = &P[l16 * PLD + lh * 8];
      pf = mk16(*(const v8bf*)p, *(const v8bf*)(p + 16));
    }
    __syncthreads();
#pragma unroll
    for (int t = 0; t < 8; ++t) {
      const bf16_t* p =
          vt + ((((size_t)b * NKV_ + kvh) * DH_) + t * 16 + l16) * (size_t)seqlen +
          kb + lh * 8;
      o[t] = wmma_bf16(pf, mk16(*(const v8bf*)p, *(const v8bf*)(p + 16)), o[t]);
    }
  }

#pragma unroll
  for (int i = 0; i < 8; ++i) {
    const float inv = 1.f / lsum[i];
#pragma unroll
    for (int t = 0; t < 8; ++t) o[t][i] *= inv;
  }
#pragma unroll
  for (int t = 0; t < 8; ++t)
#pragma unroll
    for (int i = 0; i < 8; ++i)
      out[(qrow0 + i + 8 * lh) * (size_t)(NH_ * DH_) + (size_t)h * DH_ +
          t * 16 + l16] = (bf16_t)o[t][i];
}

// ---------------- host ----------------
extern "C" void kernel_launch(void* const* d_in, const int* in_sizes, int n_in,
                              void* d_out, int out_size, void* d_ws,
                              size_t ws_size, hipStream_t stream) {
  (void)in_sizes; (void)n_in; (void)out_size; (void)ws_size;
  const float* x   = (const float*)d_in[0];
  const float* wq  = (const float*)d_in[2];
  const float* wk  = (const float*)d_in[3];
  const float* wv  = (const float*)d_in[4];
  const float* wo  = (const float*)d_in[5];
  const float* wg  = (const float*)d_in[6];
  const float* wu  = (const float*)d_in[7];
  const float* wd  = (const float*)d_in[8];
  const float* ln1 = (const float*)d_in[9];
  const float* ln2 = (const float*)d_in[10];

  char* ws = (char*)d_ws;
  size_t off = 0;
  auto alloc = [&](size_t bytes) -> char* {
    char* p = ws + off;
    off = (off + bytes + 255) & ~(size_t)255;
    return p;
  };

  // persistent region: transposed bf16 weights ([N][K]) + residual/h2
  bf16_t* wqT = (bf16_t*)alloc((size_t)DIM_ * DIM_ * 2);
  bf16_t* wkT = (bf16_t*)alloc((size_t)DIM_ * NKV_ * DH_ * 2);
  bf16_t* wvT = (bf16_t*)alloc((size_t)DIM_ * NKV_ * DH_ * 2);
  bf16_t* woT = (bf16_t*)alloc((size_t)DIM_ * DIM_ * 2);
  bf16_t* wgT = (bf16_t*)alloc((size_t)DIM_ * FFN_ * 2);
  bf16_t* wuT = (bf16_t*)alloc((size_t)DIM_ * FFN_ * 2);
  bf16_t* wdT = (bf16_t*)alloc((size_t)FFN_ * DIM_ * 2);
  float*  xmid = (float*)alloc((size_t)NTOK_ * DIM_ * 4);
  bf16_t* h2   = (bf16_t*)alloc((size_t)NTOK_ * DIM_ * 2);
  const size_t arena = off;

  // phase-1 arena
  bf16_t* h1   = (bf16_t*)alloc((size_t)NTOK_ * DIM_ * 2);
  bf16_t* q_b  = (bf16_t*)alloc((size_t)NTOK_ * NH_ * DH_ * 2);
  bf16_t* k_b  = (bf16_t*)alloc((size_t)NTOK_ * NKV_ * DH_ * 2);
  bf16_t* v_b  = (bf16_t*)alloc((size_t)NTOK_ * NKV_ * DH_ * 2);
  bf16_t* vt_b = (bf16_t*)alloc((size_t)NTOK_ * NKV_ * DH_ * 2);
  bf16_t* ao_b = (bf16_t*)alloc((size_t)NTOK_ * NH_ * DH_ * 2);

  // 1) transpose-cast weights to bf16 [N][K]
  transpose_cvt_kernel<<<dim3(DIM_ / 32, DIM_ / 32), 256, 0, stream>>>(wq, wqT, DIM_, DIM_);
  transpose_cvt_kernel<<<dim3((NKV_ * DH_) / 32, DIM_ / 32), 256, 0, stream>>>(wk, wkT, DIM_, NKV_ * DH_);
  transpose_cvt_kernel<<<dim3((NKV_ * DH_) / 32, DIM_ / 32), 256, 0, stream>>>(wv, wvT, DIM_, NKV_ * DH_);
  transpose_cvt_kernel<<<dim3(DIM_ / 32, DIM_ / 32), 256, 0, stream>>>(wo, woT, DIM_, DIM_);
  transpose_cvt_kernel<<<dim3(FFN_ / 32, DIM_ / 32), 256, 0, stream>>>(wg, wgT, DIM_, FFN_);
  transpose_cvt_kernel<<<dim3(FFN_ / 32, DIM_ / 32), 256, 0, stream>>>(wu, wuT, DIM_, FFN_);
  transpose_cvt_kernel<<<dim3(DIM_ / 32, FFN_ / 32), 256, 0, stream>>>(wd, wdT, FFN_, DIM_);

  // 2) rmsnorm1
  rmsnorm_kernel<<<NTOK_, 256, 0, stream>>>(x, ln1, h1, DIM_);

  // 3) QKV projections
  gemm_bf16_kernel<0><<<dim3(DIM_ / 256, NTOK_ / 128), 256, 0, stream>>>(
      h1, wqT, nullptr, q_b, NTOK_, NH_ * DH_, DIM_);
  gemm_bf16_kernel<0><<<dim3((NKV_ * DH_) / 256, NTOK_ / 128), 256, 0, stream>>>(
      h1, wkT, nullptr, k_b, NTOK_, NKV_ * DH_, DIM_);
  gemm_bf16_kernel<0><<<dim3((NKV_ * DH_) / 256, NTOK_ / 128), 256, 0, stream>>>(
      h1, wvT, nullptr, v_b, NTOK_, NKV_ * DH_, DIM_);

  // 4) RoPE
  {
    size_t tq = (size_t)NTOK_ * NH_ * 64;
    rope_kernel<<<(unsigned)((tq + 255) / 256), 256, 0, stream>>>(q_b, NH_, NTOK_, L_);
    size_t tk = (size_t)NTOK_ * NKV_ * 64;
    rope_kernel<<<(unsigned)((tk + 255) / 256), 256, 0, stream>>>(k_b, NKV_, NTOK_, L_);
  }

  // 5) transpose V
  {
    size_t tv = (size_t)NTOK_ * NKV_ * DH_;
    vtrans_kernel<<<(unsigned)((tv + 255) / 256), 256, 0, stream>>>(v_b, vt_b, NTOK_, L_);
  }

  // 6) attention
  attn_kernel<<<B_ * NH_ * (L_ / 16), 32, 0, stream>>>(q_b, k_b, vt_b, ao_b, L_);

  // 7) output projection + residual -> xmid (f32)
  gemm_bf16_kernel<1><<<dim3(DIM_ / 256, NTOK_ / 128), 256, 0, stream>>>(
      ao_b, woT, x, xmid, NTOK_, DIM_, NH_ * DH_);

  // 8) rmsnorm2
  rmsnorm_kernel<<<NTOK_, 256, 0, stream>>>(xmid, ln2, h2, DIM_);

  // phase-2 arena overlaps phase-1 buffers (all dead now)
  off = arena;
  bf16_t* gate = (bf16_t*)alloc((size_t)NTOK_ * FFN_ * 2);

  // 9) gate GEMM (bf16 out)
  gemm_bf16_kernel<0><<<dim3(FFN_ / 256, NTOK_ / 128), 256, 0, stream>>>(
      h2, wgT, nullptr, gate, NTOK_, FFN_, DIM_);
  // 10) up GEMM with fused silu(gate)*up, in place over gate
  gemm_bf16_kernel<2><<<dim3(FFN_ / 256, NTOK_ / 128), 256, 0, stream>>>(
      h2, wuT, gate, gate, NTOK_, FFN_, DIM_);
  // 11) down GEMM + residual -> d_out (f32)
  gemm_bf16_kernel<1><<<dim3(DIM_ / 256, NTOK_ / 128), 256, 0, stream>>>(
      gate, wdT, xmid, (float*)d_out, NTOK_, DIM_, FFN_);
}
